// GAT_27135603376491
// MI455X (gfx1250) — matrix operand order repaired and checked by reading, not compiled, not run on hardware
//
#include <hip/hip_runtime.h>

// Problem constants (from the reference)
#define NNODES 50000
#define NEDGES 800000
#define IN_CH  128
#define EE     (NEDGES + NNODES)   // edges + self loops

typedef __attribute__((ext_vector_type(16))) _Float16 v16h;
typedef __attribute__((ext_vector_type(8)))  _Float16 v8h;
typedef __attribute__((ext_vector_type(8)))  float    v8f;

// ---- order-preserving float<->uint encoding for atomicMax-based segment max ----
__device__ __forceinline__ unsigned enc_f32(float f) {
  unsigned b = __float_as_uint(f);
  return (b & 0x80000000u) ? ~b : (b | 0x80000000u);
}
__device__ __forceinline__ float dec_f32(unsigned e) {
  unsigned b = (e & 0x80000000u) ? (e & 0x7fffffffu) : ~e;
  return __uint_as_float(b);
}

// =====================================================================
// WMMA GEMM: Out[N x 64] = act(A[N x K] @ B[K x 64] + bias)
// Block = 128 threads = 4 waves; block owns a 16-row tile, wave w owns
// output cols [16w, 16w+16). K in {64,128}, compile-time -> unrolled WMMAs.
// B is staged TRANSPOSED in LDS so each lane's fragment is contiguous
// (16-byte ds_load_b128 instead of 16x ds_load_u16 + repacking).
// =====================================================================
template <int K>
__global__ __launch_bounds__(128)
void gemm_n64_wmma(const float* __restrict__ A, const float* __restrict__ B,
                   const float* __restrict__ bias, float* __restrict__ Out,
                   int act) {
  __shared__ _Float16 Alds[16][K + 8];    // row-major A tile (pitch mult. of 16B)
  __shared__ _Float16 BldsT[64][K + 8];   // B transposed: [n][k]

  const int  tid  = threadIdx.x;
  const long row0 = (long)blockIdx.x * 16;

  // Stage A tile (f32 -> f16) via float4 loads; 4-half contiguous DS stores
  {
    const int nvec = 16 * K / 4;
    for (int idx = tid; idx < nvec; idx += 128) {
      int r = idx / (K / 4), c4 = (idx % (K / 4)) * 4;
      float4 v = ((const float4*)A)[((row0 + r) * K + c4) >> 2];
      Alds[r][c4 + 0] = (_Float16)v.x;
      Alds[r][c4 + 1] = (_Float16)v.y;
      Alds[r][c4 + 2] = (_Float16)v.z;
      Alds[r][c4 + 3] = (_Float16)v.w;
    }
  }
  // Stage B (K x 64) transposed into LDS: BldsT[n][k] = B[k*64+n]
  for (int idx = tid; idx < K * 64; idx += 128) {
    int r = idx >> 6, c = idx & 63;   // consecutive tid -> consecutive c: coalesced read
    BldsT[c][r] = (_Float16)B[idx];
  }
  __syncthreads();

  const int lane = tid & 31;
  const int wave = tid >> 5;
  const int n    = (wave << 4) + (lane & 15);  // output column (0..63)
  const int m    = lane & 15;                  // A row within tile
  const int grpA = (lane >> 4) * 8;            // A-fragment K group (halves)
  const int grpB = (lane >> 4) * 8;            // B-fragment K group (halves)

  v8f acc = {};
#pragma unroll
  for (int k0 = 0; k0 < K; k0 += 32) {
    // A fragment: halves [k0+grpA, +8) and [k0+grpA+16, +8) -- two b128 loads
    const _Float16* ap = &Alds[m][k0 + grpA];
    v8h alo = *(const v8h*)ap;
    v8h ahi = *(const v8h*)(ap + 16);
    v16h a = __builtin_shufflevector(alo, ahi, 0, 1, 2, 3, 4, 5, 6, 7,
                                               8, 9, 10, 11, 12, 13, 14, 15);
    // B fragment: halves [k0+grpB, +8) and [k0+grpB+16, +8) -- two b128 loads
    const _Float16* bp = &BldsT[n][k0 + grpB];
    v8h blo = *(const v8h*)bp;
    v8h bhi = *(const v8h*)(bp + 16);
    v16h b = __builtin_shufflevector(blo, bhi, 0, 1, 2, 3, 4, 5, 6, 7,
                                               8, 9, 10, 11, 12, 13, 14, 15);
    // (neg_a, A, neg_b, B, c_mod, C, reuse_a, reuse_b)
    acc = __builtin_amdgcn_wmma_f32_16x16x32_f16(false, a, false, b,
                                                 (short)0, acc, false, false);
  }

  const float bv    = bias ? bias[n] : 0.0f;
  const int   mbase = (lane >> 4) * 8;
#pragma unroll
  for (int r = 0; r < 8; ++r) {
    float v = acc[r] + bv;
    if (act == 1) v = fmaxf(v, 0.0f);  // ReLU epilogue (projector layer 1)
    Out[(row0 + mbase + r) * 64 + n] = v;
  }
}

// =====================================================================
// Per-node attention dot products: aS/aD[i,h] = <h[i, h*C : h*C+C], att>
// =====================================================================
__global__ void att_dots(const float* __restrict__ h, const float* __restrict__ attS,
                         const float* __restrict__ attD, float* __restrict__ aS,
                         float* __restrict__ aD, int H) {
  int i = blockIdx.x * blockDim.x + threadIdx.x;
  if (i >= NNODES) return;
  const float* hr = h + (long)i * 64;
  const int C = 64 / H;
  for (int hd = 0; hd < H; ++hd) {
    float s = 0.0f, d = 0.0f;
    for (int c = 0; c < C; ++c) {
      float v = hr[hd * C + c];
      s += v * attS[hd * C + c];
      d += v * attD[hd * C + c];
    }
    aS[i * H + hd] = s;
    aD[i * H + hd] = d;
  }
}

// =====================================================================
// Pass 1: leaky-relu logits + per-destination segment max (encoded atomicMax)
// =====================================================================
__global__ void edge_logit_max(const long long* __restrict__ ei,
                               const float* __restrict__ aS, const float* __restrict__ aD,
                               float* __restrict__ logit, unsigned* __restrict__ mEnc,
                               int H) {
  long idx = (long)blockIdx.x * blockDim.x + threadIdx.x;
  if (idx >= (long)EE * H) return;
  int  i  = (int)(idx / H);
  int  hd = (int)(idx % H);
  long s  = (i < NEDGES) ? ei[i] : (long)(i - NEDGES);
  long d  = (i < NEDGES) ? ei[NEDGES + i] : (long)(i - NEDGES);
  float e = aS[s * H + hd] + aD[d * H + hd];
  e = (e > 0.0f) ? e : 0.2f * e;  // leaky_relu, slope 0.2
  logit[idx] = e;
  atomicMax(&mEnc[d * H + hd], enc_f32(e));
}

// =====================================================================
// Pass 2: p = exp(logit - max); accumulate softmax denominator
// =====================================================================
__global__ void edge_exp_den(const long long* __restrict__ ei,
                             float* __restrict__ logit,
                             const unsigned* __restrict__ mEnc,
                             float* __restrict__ den, int H) {
  long idx = (long)blockIdx.x * blockDim.x + threadIdx.x;
  if (idx >= (long)EE * H) return;
  int  i  = (int)(idx / H);
  int  hd = (int)(idx % H);
  long d  = (i < NEDGES) ? ei[NEDGES + i] : (long)(i - NEDGES);
  float m = dec_f32(mEnc[d * H + hd]);
  float p = __expf(logit[idx] - m);
  logit[idx] = p;
  atomicAdd(&den[d * H + hd], p);
}

// =====================================================================
// Pass 3: one wave per edge; scatter-add alpha * h[src] into agg[dst]
// (64 channels -> 2 per lane). Accumulators live in L2 (12.8 MB << 192 MB).
// =====================================================================
__global__ void edge_scatter(const long long* __restrict__ ei,
                             const float* __restrict__ h,
                             const float* __restrict__ logit,
                             const float* __restrict__ den,
                             float* __restrict__ agg, int H) {
  int w    = (blockIdx.x * blockDim.x + threadIdx.x) >> 5;
  int lane = threadIdx.x & 31;
  if (w >= EE) return;
  long s = (w < NEDGES) ? ei[w] : (long)(w - NEDGES);
  long d = (w < NEDGES) ? ei[NEDGES + w] : (long)(w - NEDGES);
  const int C = 64 / H;
#pragma unroll
  for (int c = lane; c < 64; c += 32) {
    int   hd    = c / C;
    float alpha = logit[(long)w * H + hd] / (den[d * H + hd] + 1e-16f);
    atomicAdd(&agg[d * 64 + c], h[s * 64 + c] * alpha);
  }
}

// =====================================================================
// Elementwise: out = act(in + bias[col]); mode 0=none, 1=relu, 2=elu
// =====================================================================
__global__ void bias_act64(const float* __restrict__ in, const float* __restrict__ bias,
                           float* __restrict__ out, int mode) {
  long idx = (long)blockIdx.x * blockDim.x + threadIdx.x;
  if (idx >= (long)NNODES * 64) return;
  int   col = (int)(idx & 63);
  float v   = in[idx] + bias[col];
  if (mode == 1)      v = fmaxf(v, 0.0f);
  else if (mode == 2) v = (v > 0.0f) ? v : (__expf(v) - 1.0f);
  out[idx] = v;
}

// =====================================================================
// Host-side launch sequence
// =====================================================================
extern "C" void kernel_launch(void* const* d_in, const int* in_sizes, int n_in,
                              void* d_out, int out_size, void* d_ws, size_t ws_size,
                              hipStream_t stream) {
  const float*     x   = (const float*)d_in[0];
  const long long* ei  = (const long long*)d_in[1];  // int64 edge_index [2,E]
  const float*     W1  = (const float*)d_in[2];
  const float*     aS1 = (const float*)d_in[3];
  const float*     aD1 = (const float*)d_in[4];
  const float*     b1  = (const float*)d_in[5];
  const float*     W2  = (const float*)d_in[6];
  const float*     aS2 = (const float*)d_in[7];
  const float*     aD2 = (const float*)d_in[8];
  const float*     b2  = (const float*)d_in[9];
  const float*     pw1 = (const float*)d_in[10];
  const float*     pb1 = (const float*)d_in[11];
  const float*     pw2 = (const float*)d_in[12];
  const float*     pb2 = (const float*)d_in[13];

  // Workspace layout (floats): 3x N*64 ping-pong + attention scalars + logits
  float*    buf0  = (float*)d_ws;                          // h1 / elu(agg1+b1) / agg2+b2
  float*    buf1  = buf0 + (size_t)NNODES * 64;            // aggregation accumulator
  float*    buf2  = buf1 + (size_t)NNODES * 64;            // h2 / relu-proj
  float*    aS    = buf2 + (size_t)NNODES * 64;            // [N*2]
  float*    aD    = aS + (size_t)NNODES * 2;               // [N*2]
  float*    den   = aD + (size_t)NNODES * 2;               // [N*2]
  unsigned* mEnc  = (unsigned*)(den + (size_t)NNODES * 2); // [N*2]
  float*    logit = (float*)(mEnc + (size_t)NNODES * 2);   // [EE*2]

  const int gemmGrid = NNODES / 16;  // 3125 (exact)
  const int nodeGrid = (NNODES + 255) / 256;
  const int ewGrid   = ((long)NNODES * 64 + 255) / 256;
  const int scatGrid = (EE + 7) / 8;  // 8 waves / 256-thread block

  // ---------------- conv1 (heads=2, hid=32) ----------------
  gemm_n64_wmma<128><<<gemmGrid, 128, 0, stream>>>(x, W1, nullptr, buf0, 0);
  att_dots<<<nodeGrid, 256, 0, stream>>>(buf0, aS1, aD1, aS, aD, 2);
  (void)hipMemsetAsync(mEnc, 0, (size_t)NNODES * 2 * sizeof(unsigned), stream);
  (void)hipMemsetAsync(den,  0, (size_t)NNODES * 2 * sizeof(float), stream);
  (void)hipMemsetAsync(buf1, 0, (size_t)NNODES * 64 * sizeof(float), stream);
  {
    long tot = (long)EE * 2;
    int  g   = (int)((tot + 255) / 256);
    edge_logit_max<<<g, 256, 0, stream>>>(ei, aS, aD, logit, mEnc, 2);
    edge_exp_den<<<g, 256, 0, stream>>>(ei, logit, mEnc, den, 2);
  }
  edge_scatter<<<scatGrid, 256, 0, stream>>>(ei, buf0, logit, den, buf1, 2);
  bias_act64<<<ewGrid, 256, 0, stream>>>(buf1, b1, buf0, 2);  // ELU

  // ---------------- conv2 (heads=1, out=64) ----------------
  gemm_n64_wmma<64><<<gemmGrid, 128, 0, stream>>>(buf0, W2, nullptr, buf2, 0);
  att_dots<<<nodeGrid, 256, 0, stream>>>(buf2, aS2, aD2, aS, aD, 1);
  (void)hipMemsetAsync(mEnc, 0, (size_t)NNODES * sizeof(unsigned), stream);
  (void)hipMemsetAsync(den,  0, (size_t)NNODES * sizeof(float), stream);
  (void)hipMemsetAsync(buf1, 0, (size_t)NNODES * 64 * sizeof(float), stream);
  {
    int g = (EE + 255) / 256;
    edge_logit_max<<<g, 256, 0, stream>>>(ei, aS, aD, logit, mEnc, 1);
    edge_exp_den<<<g, 256, 0, stream>>>(ei, logit, mEnc, den, 1);
  }
  edge_scatter<<<scatGrid, 256, 0, stream>>>(ei, buf2, logit, den, buf1, 1);
  bias_act64<<<ewGrid, 256, 0, stream>>>(buf1, b2, buf0, 0);  // + b2

  // ---------------- projector MLP ----------------
  gemm_n64_wmma<64><<<gemmGrid, 128, 0, stream>>>(buf0, pw1, pb1, buf2, 1);          // ReLU
  gemm_n64_wmma<64><<<gemmGrid, 128, 0, stream>>>(buf2, pw2, pb2, (float*)d_out, 0); // final
}
